// omics1_encoder_84851373899829
// MI455X (gfx1250) — matrix operand order repaired and checked by reading, not compiled, not run on hardware
//
#include <hip/hip_runtime.h>
#include <hip/hip_bf16.h>

// Problem constants (match reference)
#define N_NODES 8192
#define N_EDGES (8192 * 32)
#define DIN 3000
#define HID 256
#define DOUTD 64

typedef __attribute__((ext_vector_type(16))) __bf16 v16bf;
typedef __attribute__((ext_vector_type(8)))  float  v8f;

// ---------------- degree / symmetric norm ----------------
__global__ void k_deg_init(float* deg) {
  int i = blockIdx.x * blockDim.x + threadIdx.x;
  if (i < N_NODES) deg[i] = 1.0f;  // self-loop
}
__global__ void k_deg_accum(const int* __restrict__ dst, float* deg) {
  int e = blockIdx.x * blockDim.x + threadIdx.x;
  if (e < N_EDGES) atomicAdd(&deg[dst[e]], 1.0f);
}
__global__ void k_dinv(float* deg) {
  int i = blockIdx.x * blockDim.x + threadIdx.x;
  if (i < N_NODES) deg[i] = rsqrtf(deg[i]);  // deg >= 1 always
}

// ---------------- bf16 WMMA GEMM ----------------
// C[M,Nc] = (opt relu)A[M,K] x B[K,Nc], f32 accumulation.
// Block = NWAVES waves sharing one staged A tile (16x32 fp32 in LDS).
// Each wave computes 16 x (16*NTILES) of C. All NTILES B fragments are built
// BEFORE the NTILES WMMAs so the matrix ops issue back-to-back with no
// WAR hazard NOPs; LDS tiles are double-buffered so next-tile global loads
// overlap the current tile's WMMAs.

template <int NTILES, int NCOLS>
__device__ __forceinline__ void wmma_step(const float* sA, const float* sB,
                                          int lane, int wave, v8f* acc)
{
  const int half = lane >> 4, l15 = lane & 15;
  // A 16x32 bf16 fragment: lane<16 -> K {0..7,16..23}; lane>=16 -> +8
  v16bf va;
#pragma unroll
  for (int v = 0; v < 16; ++v) {
    int ka = (v & 7) + ((v >> 3) << 4) + (half << 3);
    va[v] = (__bf16)sA[l15 * 32 + ka];
  }
  // Build ALL B fragments first (B 32x16: lane<16 -> K {0..7,16..23}; lane>=16 -> +8)
  v16bf vb[NTILES];
#pragma unroll
  for (int nt = 0; nt < NTILES; ++nt) {
    int cb = (wave * NTILES + nt) * 16;
#pragma unroll
    for (int v = 0; v < 16; ++v) {
      int kB = (v & 7) + ((v >> 3) << 4) + (half << 3);
      vb[nt][v] = (__bf16)sB[kB * NCOLS + cb + l15];
    }
  }
  // Back-to-back WMMAs: disjoint B/acc registers -> no inter-WMMA hazards
#pragma unroll
  for (int nt = 0; nt < NTILES; ++nt)
    acc[nt] = __builtin_amdgcn_wmma_f32_16x16x32_bf16(false, va, false, vb[nt],
                                                      (short)0, acc[nt],
                                                      false, false);
}

template <int NWAVES, int NTILES>
__global__ __launch_bounds__(32 * NWAVES)
void k_wmma_gemm(const float* __restrict__ A, const float* __restrict__ B,
                 float* __restrict__ C, int Nc, int K, int reluA)
{
  constexpr int NCOLS = 16 * NTILES * NWAVES;  // block tile width
  constexpr int NT = 32 * NWAVES;              // threads per block
  __shared__ float sA[2][16 * 32];
  __shared__ float sB[2][32 * NCOLS];
  const int tid = threadIdx.x;
  const int lane = tid & 31, wave = tid >> 5;
  const int half = lane >> 4, l15 = lane & 15;
  const long tm = blockIdx.x;
  const long colBase = (long)blockIdx.y * NCOLS;

  v8f acc[NTILES] = {};

  auto stage = [&](int it, int buf) {
    const int kb = it * 32;
    if (kb + 32 <= K) {  // uniform: fast path, no bounds checks
      for (int t = tid; t < 16 * 32; t += NT) {
        int r = t >> 5, c = t & 31;
        float a = A[(tm * 16 + r) * (long)K + kb + c];
        if (reluA) a = fmaxf(a, 0.0f);
        sA[buf][t] = a;
      }
      for (int t = tid; t < 32 * NCOLS; t += NT) {
        int r = t / NCOLS, c = t % NCOLS;  // NCOLS is power of two
        sB[buf][t] = B[(long)(kb + r) * Nc + colBase + c];
      }
    } else {  // zero-padded tail (only final tile when K % 32 != 0)
      for (int t = tid; t < 16 * 32; t += NT) {
        int r = t >> 5, c = t & 31;
        float a = (kb + c < K) ? A[(tm * 16 + r) * (long)K + kb + c] : 0.0f;
        if (reluA) a = fmaxf(a, 0.0f);
        sA[buf][t] = a;
      }
      for (int t = tid; t < 32 * NCOLS; t += NT) {
        int r = t / NCOLS, c = t % NCOLS;
        sB[buf][t] = (kb + r < K) ? B[(long)(kb + r) * Nc + colBase + c] : 0.0f;
      }
    }
  };

  const int KT = (K + 31) >> 5;
  stage(0, 0);
  __syncthreads();
  for (int it = 0; it < KT; ++it) {
    if (it + 1 < KT) stage(it + 1, (it + 1) & 1);  // prefetch next tile
    wmma_step<NTILES, NCOLS>(sA[it & 1], sB[it & 1], lane, wave, acc);
    __syncthreads();  // staged buf^1 complete AND buf consumed
  }

  // C layout: lanes 0-15 hold M=r (vgpr r), lanes 16-31 hold M=8+r
#pragma unroll
  for (int nt = 0; nt < NTILES; ++nt) {
#pragma unroll
    for (int r = 0; r < 8; ++r) {
      long row = tm * 16 + half * 8 + r;
      C[row * (long)Nc + colBase + (wave * NTILES + nt) * 16 + l15] = acc[nt][r];
    }
  }
}

// ---------------- GCN aggregation: out = D^-1/2 (A+I) D^-1/2 (xw) + b ------
template <int F>
__global__ void k_agg_init(const float* __restrict__ xw,
                           const float* __restrict__ dinv,
                           const float* __restrict__ bias,
                           float* __restrict__ out)
{
  long i = (long)blockIdx.x * blockDim.x + threadIdx.x;
  if (i >= (long)N_NODES * F) return;
  int node = (int)(i / F), f = (int)(i % F);
  float d = dinv[node];
  out[i] = xw[i] * d * d + bias[f];  // self-loop contribution + bias
}

template <int F>
__global__ void k_agg_edges(const float* __restrict__ xw,
                            const int* __restrict__ src,
                            const int* __restrict__ dst,
                            const float* __restrict__ dinv,
                            float* out)
{
  long i = (long)blockIdx.x * blockDim.x + threadIdx.x;
  if (i >= (long)N_EDGES * F) return;
  int e = (int)(i / F), f = (int)(i % F);
  int s = src[e], d = dst[e];
  float w = dinv[s] * dinv[d];
  atomicAdd(&out[(long)d * F + f], xw[(long)s * F + f] * w);
}

// ---------------- readout helpers ----------------
__global__ __launch_bounds__(256)
void k_rowsum(const float* __restrict__ mask, float* __restrict__ rs)
{
  __shared__ float sm[256];
  int row = blockIdx.x;
  float s = 0.0f;
  for (int c = threadIdx.x; c < N_NODES; c += 256)
    s += mask[(long)row * N_NODES + c];
  sm[threadIdx.x] = s;
  __syncthreads();
  for (int off = 128; off > 0; off >>= 1) {
    if (threadIdx.x < off) sm[threadIdx.x] += sm[threadIdx.x + off];
    __syncthreads();
  }
  if (threadIdx.x == 0) rs[row] = sm[0];
}

__global__ void k_gsig(const float* __restrict__ vsum,
                       const float* __restrict__ rs,
                       float* __restrict__ g)
{
  int i = blockIdx.x * blockDim.x + threadIdx.x;
  if (i >= N_NODES) return;
  float inv = 1.0f / rs[i];
  float v[DOUTD];
  float nrm = 0.0f;
  for (int d = 0; d < DOUTD; ++d) {
    float x = vsum[(long)i * DOUTD + d] * inv;
    v[d] = x;
    nrm += x * x;
  }
  float s = 1.0f / fmaxf(sqrtf(nrm), 1e-12f);
  for (int d = 0; d < DOUTD; ++d)
    g[(long)i * DOUTD + d] = 1.0f / (1.0f + expf(-v[d] * s));
}

// ---------------- bilinear discriminator ----------------
__global__ void k_disc(const float* __restrict__ c, const float* __restrict__ hpl,
                       const float* __restrict__ hmi, const float* __restrict__ Wb,
                       const float* __restrict__ bb, float* __restrict__ ret)
{
  int i = blockIdx.x * blockDim.x + threadIdx.x;
  if (i >= N_NODES) return;
  float sc1 = bb[0], sc2 = bb[0];
  for (int d = 0; d < DOUTD; ++d) {
    float t = 0.0f;
    for (int e = 0; e < DOUTD; ++e)
      t += c[(long)i * DOUTD + e] * Wb[d * DOUTD + e];  // cW = c @ Wb^T
    sc1 += hpl[(long)i * DOUTD + d] * t;
    sc2 += hmi[(long)i * DOUTD + d] * t;
  }
  ret[2 * i + 0] = sc1;
  ret[2 * i + 1] = sc2;
}

// ---------------- driver ----------------
extern "C" void kernel_launch(void* const* d_in, const int* in_sizes, int n_in,
                              void* d_out, int out_size, void* d_ws, size_t ws_size,
                              hipStream_t stream)
{
  (void)in_sizes; (void)n_in; (void)out_size; (void)ws_size;
  const float* feat   = (const float*)d_in[0];
  const float* feat_a = (const float*)d_in[1];
  const float* mask   = (const float*)d_in[2];
  const float* W1     = (const float*)d_in[3];
  const float* b1     = (const float*)d_in[4];
  const float* W2     = (const float*)d_in[5];
  const float* b2     = (const float*)d_in[6];
  const float* Wb     = (const float*)d_in[7];
  const float* bb     = (const float*)d_in[8];
  const int*   ei     = (const int*)d_in[9];
  const int* src = ei;            // edge_index[0]
  const int* dst = ei + N_EDGES;  // edge_index[1]

  float* out_emb   = (float*)d_out;                        // [N, 64]
  float* out_ret   = out_emb + (long)N_NODES * DOUTD;      // [N, 2]
  float* out_ret_a = out_ret + (long)N_NODES * 2;          // [N, 2]

  // carve workspace (~27 MB)
  char* wp = (char*)d_ws;
  auto carve = [&](size_t bytes) -> void* {
    void* p = (void*)wp;
    wp += (bytes + 255) & ~(size_t)255;
    return p;
  };
  float* dinv   = (float*)carve((size_t)N_NODES * 4);
  float* xw     = (float*)carve((size_t)N_NODES * HID * 4);
  float* z      = (float*)carve((size_t)N_NODES * HID * 4);
  float* zw     = (float*)carve((size_t)N_NODES * DOUTD * 4);
  float* emb_a  = (float*)carve((size_t)N_NODES * DOUTD * 4);
  float* rowsum = (float*)carve((size_t)N_NODES * 4);
  float* vsum   = (float*)carve((size_t)N_NODES * DOUTD * 4);
  float* g      = (float*)carve((size_t)N_NODES * DOUTD * 4);
  float* g_a    = (float*)carve((size_t)N_NODES * DOUTD * 4);

  // --- GCN symmetric normalization ---
  k_deg_init<<<N_NODES / 256, 256, 0, stream>>>(dinv);
  k_deg_accum<<<N_EDGES / 256, 256, 0, stream>>>(dst, dinv);
  k_dinv<<<N_NODES / 256, 256, 0, stream>>>(dinv);

  auto encode = [&](const float* x, float* emb) {
    // layer 1: x @ W1, block tile 16x256 (full Nc) -> x read exactly once
    k_wmma_gemm<4, 4><<<dim3(N_NODES / 16, 1), 128, 0, stream>>>(x, W1, xw, HID, DIN, 0);
    k_agg_init<HID><<<(N_NODES * HID) / 256, 256, 0, stream>>>(xw, dinv, b1, z);
    k_agg_edges<HID><<<(N_EDGES * HID) / 256, 256, 0, stream>>>(xw, src, dst, dinv, z);
    // layer 2: relu fused into A-load of GEMM; block tile 16x64 (full Nc)
    k_wmma_gemm<4, 1><<<dim3(N_NODES / 16, 1), 128, 0, stream>>>(z, W2, zw, DOUTD, HID, 1);
    k_agg_init<DOUTD><<<(N_NODES * DOUTD) / 256, 256, 0, stream>>>(zw, dinv, b2, emb);
    k_agg_edges<DOUTD><<<(N_EDGES * DOUTD) / 256, 256, 0, stream>>>(zw, src, dst, dinv, emb);
  };
  encode(feat, out_emb);
  encode(feat_a, emb_a);

  // --- readout: g = sigmoid(normalize(mask @ emb / rowsum)) ---
  k_rowsum<<<N_NODES, 256, 0, stream>>>(mask, rowsum);
  k_wmma_gemm<4, 1><<<dim3(N_NODES / 16, 1), 128, 0, stream>>>(mask, out_emb, vsum, DOUTD, N_NODES, 0);
  k_gsig<<<N_NODES / 256, 256, 0, stream>>>(vsum, rowsum, g);
  k_wmma_gemm<4, 1><<<dim3(N_NODES / 16, 1), 128, 0, stream>>>(mask, emb_a, vsum, DOUTD, N_NODES, 0);
  k_gsig<<<N_NODES / 256, 256, 0, stream>>>(vsum, rowsum, g_a);

  // --- bilinear discriminator ---
  k_disc<<<N_NODES / 256, 256, 0, stream>>>(g, out_emb, emb_a, Wb, bb, out_ret);
  k_disc<<<N_NODES / 256, 256, 0, stream>>>(g_a, emb_a, out_emb, Wb, bb, out_ret_a);
}
